// GNN_70300024701460
// MI455X (gfx1250) — matrix operand order repaired and checked by reading, not compiled, not run on hardware
//
#include <hip/hip_runtime.h>

// ---------------------------------------------------------------------------
// 2-layer GATConv (heads=1, self-loops with fill='mean') for MI455X / gfx1250.
// N=50000 nodes, E=800000 edges, all feature dims = 128.
//
// Optimization strategy (see analysis): fold (ea@We)@ae -> ea@(We@ae) so the
// only heavy HBM traffic is one pass over edge_attr (410MB, both layers at
// once). Dense h=x@W uses v_wmma_f32_16x16x32_f16 (f32 accum). Softmax +
// scatter are L2-resident atomics (per-node arrays ~25MB << 192MB L2).
// ---------------------------------------------------------------------------

#define NN 50000
#define EE 800000
#define CC 128
#define TT (EE + NN)          // edges + self loops
#define NTILES (NN / 16)      // 3125 row tiles for WMMA

typedef __attribute__((ext_vector_type(16))) _Float16 v16h;
typedef __attribute__((ext_vector_type(8)))  float    v8f;

// ---- order-preserving float<->uint encoding for atomicMax on floats -------
__device__ __forceinline__ unsigned fenc(float f) {
  unsigned u = __float_as_uint(f);
  return (u & 0x80000000u) ? ~u : (u | 0x80000000u);
}
__device__ __forceinline__ float fdec(unsigned u) {
  return (u & 0x80000000u) ? __uint_as_float(u ^ 0x80000000u)
                           : __uint_as_float(~u);
}

__device__ __forceinline__ float wave_sum(float v) {
  #pragma unroll
  for (int off = 16; off; off >>= 1) v += __shfl_xor(v, off, 32);
  return v;
}

// ---- v_e = We @ att_e for both layers (128 threads, 1 block) --------------
__global__ void k_vec(const float* We1, const float* ae1,
                      const float* We2, const float* ae2,
                      float* ve1, float* ve2) {
  int i = threadIdx.x;
  float s1 = 0.f, s2 = 0.f;
  for (int j = 0; j < CC; ++j) {
    s1 += We1[i * CC + j] * ae1[j];
    s2 += We2[i * CC + j] * ae2[j];
  }
  ve1[i] = s1; ve2[i] = s2;
}

// ---- pack W (f32 row-major) into wave32 WMMA B-fragment order (f16) -------
// b[i] for lane l in (ct,kc) tile must equal W[kc*32 + (i<8 ? (l>>4)*8 + i : 16 + (l>>4)*8 + i-8)][ct*16 + (l&15)]
__global__ void k_pack(const float* W, _Float16* Wpk) {
  int idx = blockIdx.x * blockDim.x + threadIdx.x;
  if (idx >= 16384) return;
  int i = idx & 15, lane = (idx >> 4) & 31, kc = (idx >> 9) & 3, ct = idx >> 11;
  int row = kc * 32 + (i < 8 ? (lane >> 4) * 8 + i : 16 + (lane >> 4) * 8 + (i - 8));
  int col = ct * 16 + (lane & 15);
  Wpk[idx] = (_Float16)W[row * CC + col];
}

__global__ void k_zero_pre(float* deg, float* es1, float* es2) {
  int n = blockIdx.x * blockDim.x + threadIdx.x;
  if (n < NN) { deg[n] = 0.f; es1[n] = 0.f; es2[n] = 0.f; }
}

// ---- one pass over edge_attr: per-edge e.v_e1 / e.v_e2, deg, segment sums -
__global__ void k_edge_dot(const float* __restrict__ ea, const long long* __restrict__ ei,
                           const float* ve1, const float* ve2,
                           float* ev1, float* ev2,
                           float* deg, float* es1, float* es2) {
  __shared__ __align__(16) float sv[256];
  int tid = threadIdx.x;
  sv[tid] = (tid < 128) ? ve1[tid] : ve2[tid - 128];
  __syncthreads();
  int lane = tid & 31, w = tid >> 5;
  int e = blockIdx.x * 8 + w;                       // grid sized exactly E/8
  const float4 a  = *(const float4*)(ea + (size_t)e * CC + lane * 4);
  const float4 b1 = *(const float4*)(sv + lane * 4);
  const float4 b2 = *(const float4*)(sv + 128 + lane * 4);
  float d1 = a.x * b1.x + a.y * b1.y + a.z * b1.z + a.w * b1.w;
  float d2 = a.x * b2.x + a.y * b2.y + a.z * b2.z + a.w * b2.w;
  d1 = wave_sum(d1); d2 = wave_sum(d2);
  if (lane == 0) {
    ev1[e] = d1; ev2[e] = d2;
    int d = (int)ei[EE + e];
    atomicAdd(deg + d, 1.0f);
    atomicAdd(es1 + d, d1);
    atomicAdd(es2 + d, d2);
  }
}

// ---- H = X @ W via WMMA f16->f32: one wave per 16-row stripe --------------
__global__ __launch_bounds__(128)
void k_gemm(const float* __restrict__ X, const _Float16* __restrict__ Wpk,
            float* __restrict__ H) {
  int lane = threadIdx.x;                 // 0..31
  int t = blockIdx.x * blockDim.y + threadIdx.y;
  if (t >= NTILES) return;
  int m16 = lane & 15, hi = lane >> 4;
  const float* xrow = X + (size_t)(t * 16 + m16) * CC;
  const v16h* wp = (const v16h*)Wpk;

  v8f acc[8];
  #pragma unroll
  for (int ct = 0; ct < 8; ++ct)
    #pragma unroll
    for (int j = 0; j < 8; ++j) acc[ct][j] = 0.0f;

  #pragma unroll
  for (int kc = 0; kc < 4; ++kc) {
    int k0 = kc * 32;
    // A-frag (16x32 f16): lane<16 -> K {k0..k0+7, k0+16..k0+23}; lane>=16 shifts by 8
    float4 f0 = *(const float4*)(xrow + k0 + hi * 8);
    float4 f1 = *(const float4*)(xrow + k0 + hi * 8 + 4);
    float4 f2 = *(const float4*)(xrow + k0 + 16 + hi * 8);
    float4 f3 = *(const float4*)(xrow + k0 + 16 + hi * 8 + 4);
    v16h a;
    a[0]=(_Float16)f0.x;  a[1]=(_Float16)f0.y;  a[2]=(_Float16)f0.z;  a[3]=(_Float16)f0.w;
    a[4]=(_Float16)f1.x;  a[5]=(_Float16)f1.y;  a[6]=(_Float16)f1.z;  a[7]=(_Float16)f1.w;
    a[8]=(_Float16)f2.x;  a[9]=(_Float16)f2.y;  a[10]=(_Float16)f2.z; a[11]=(_Float16)f2.w;
    a[12]=(_Float16)f3.x; a[13]=(_Float16)f3.y; a[14]=(_Float16)f3.z; a[15]=(_Float16)f3.w;
    #pragma unroll
    for (int ct = 0; ct < 8; ++ct) {
      v16h b = wp[(ct * 4 + kc) * 32 + lane];   // pre-swizzled B fragment
      acc[ct] = __builtin_amdgcn_wmma_f32_16x16x32_f16(
          false, a, false, b, (short)0, acc[ct], false, false);
    }
  }
  // C/D layout: VGPR j -> M = j + 8*hi ; N = lane&15
  #pragma unroll
  for (int ct = 0; ct < 8; ++ct)
    #pragma unroll
    for (int j = 0; j < 8; ++j)
      H[(size_t)(t * 16 + j + hi * 8) * CC + ct * 16 + m16] = acc[ct][j];
}

// ---- a_s[n] = h[n].att_s ; a_d[n] = h[n].att_d (wave per node) ------------
__global__ void k_asad(const float* __restrict__ H, const float* atts, const float* attd,
                       float* as_, float* ad_) {
  int lane = threadIdx.x & 31, w = threadIdx.x >> 5;
  int n = blockIdx.x * 8 + w;
  if (n >= NN) return;
  float4 h4 = *(const float4*)(H + (size_t)n * CC + lane * 4);
  float4 s4 = *(const float4*)(atts + lane * 4);
  float4 d4 = *(const float4*)(attd + lane * 4);
  float ds = h4.x * s4.x + h4.y * s4.y + h4.z * s4.z + h4.w * s4.w;
  float dd = h4.x * d4.x + h4.y * d4.y + h4.z * d4.z + h4.w * d4.w;
  ds = wave_sum(ds); dd = wave_sum(dd);
  if (lane == 0) { as_[n] = ds; ad_[n] = dd; }
}

// ---- per-layer init: out = bias, softmax stats cleared --------------------
__global__ void k_init_layer(float* out, const float* b, unsigned* menc, float* ssum) {
  int idx = blockIdx.x * blockDim.x + threadIdx.x;
  if (idx < NN * CC) out[idx] = b[idx & (CC - 1)];
  if (idx < NN) { menc[idx] = 0u; ssum[idx] = 0.f; }   // 0u == below every float code
}

// ---- alpha = leaky_relu(a_s[src]+a_d[dst]+a_e), segment max ---------------
__global__ void k_alpha(const long long* __restrict__ ei, const float* __restrict__ ev,
                        const float* es, const float* deg,
                        const float* __restrict__ as_, const float* __restrict__ ad_,
                        float* alph, unsigned* menc) {
  int t = blockIdx.x * blockDim.x + threadIdx.x;
  if (t >= TT) return;
  int s, d; float av;
  if (t < EE) { s = (int)ei[t]; d = (int)ei[EE + t]; av = ev[t]; }
  else        { int n = t - EE; s = n; d = n; av = es[n] / fmaxf(deg[n], 1.0f); }
  float z = as_[s] + ad_[d] + av;
  float lr = z > 0.f ? z : 0.2f * z;
  alph[t] = lr;
  atomicMax(menc + d, fenc(lr));
}

// ---- ex = exp(alpha - m[dst]), segment sum --------------------------------
__global__ void k_expsum(const long long* __restrict__ ei, const unsigned* __restrict__ menc,
                         float* alph, float* ssum) {
  int t = blockIdx.x * blockDim.x + threadIdx.x;
  if (t >= TT) return;
  int d = (t < EE) ? (int)ei[EE + t] : (t - EE);
  float ex = expf(alph[t] - fdec(menc[d]));
  alph[t] = ex;
  atomicAdd(ssum + d, ex);
}

// ---- out[dst] += (ex/ssum[dst]) * h[src]  (wave per edge, L2 atomics) -----
__global__ void k_scatter(const long long* __restrict__ ei, const float* __restrict__ alph,
                          const float* __restrict__ ssum, const float* __restrict__ H,
                          float* out) {
  int lane = threadIdx.x & 31, w = threadIdx.x >> 5;
  int t = blockIdx.x * 8 + w;
  if (t >= TT) return;
  int s, d;
  if (t < EE) { s = (int)ei[t]; d = (int)ei[EE + t]; }
  else        { s = t - EE; d = s; }
  float wgt = alph[t] / ssum[d];
  float4 hv = *(const float4*)(H + (size_t)s * CC + lane * 4);
  float* o = out + (size_t)d * CC + lane * 4;
  atomicAdd(o + 0, wgt * hv.x);
  atomicAdd(o + 1, wgt * hv.y);
  atomicAdd(o + 2, wgt * hv.z);
  atomicAdd(o + 3, wgt * hv.w);
}

__global__ void k_relu(float* p) {
  int idx = blockIdx.x * blockDim.x + threadIdx.x;
  if (idx < NN * CC) p[idx] = fmaxf(p[idx], 0.f);
}

// ---------------------------------------------------------------------------
static inline char* carve(char*& p, size_t bytes) {
  char* r = p;
  p += (bytes + 255) & ~(size_t)255;
  return r;
}

extern "C" void kernel_launch(void* const* d_in, const int* in_sizes, int n_in,
                              void* d_out, int out_size, void* d_ws, size_t ws_size,
                              hipStream_t stream) {
  (void)in_sizes; (void)n_in; (void)out_size; (void)ws_size;
  const float*     x   = (const float*)d_in[0];
  const long long* ei  = (const long long*)d_in[1];   // int64 [2,E]: src then dst
  const float*     ea  = (const float*)d_in[2];
  const float*     W1  = (const float*)d_in[3];
  const float*     a1s = (const float*)d_in[4];
  const float*     a1d = (const float*)d_in[5];
  const float*     We1 = (const float*)d_in[6];
  const float*     ae1 = (const float*)d_in[7];
  const float*     b1  = (const float*)d_in[8];
  const float*     W2  = (const float*)d_in[9];
  const float*     a2s = (const float*)d_in[10];
  const float*     a2d = (const float*)d_in[11];
  const float*     We2 = (const float*)d_in[12];
  const float*     ae2 = (const float*)d_in[13];
  const float*     b2  = (const float*)d_in[14];
  float* out = (float*)d_out;

  // workspace carve (~63 MB total)
  char* p = (char*)d_ws;
  float*    h     = (float*)carve(p, (size_t)NN * CC * 4);
  float*    o1    = (float*)carve(p, (size_t)NN * CC * 4);
  float*    ev1   = (float*)carve(p, (size_t)EE * 4);
  float*    ev2   = (float*)carve(p, (size_t)EE * 4);
  float*    alph  = (float*)carve(p, (size_t)TT * 4);
  float*    deg   = (float*)carve(p, (size_t)NN * 4);
  float*    es1   = (float*)carve(p, (size_t)NN * 4);
  float*    es2   = (float*)carve(p, (size_t)NN * 4);
  float*    as_   = (float*)carve(p, (size_t)NN * 4);
  float*    ad_   = (float*)carve(p, (size_t)NN * 4);
  unsigned* menc  = (unsigned*)carve(p, (size_t)NN * 4);
  float*    ssum  = (float*)carve(p, (size_t)NN * 4);
  float*    ve1   = (float*)carve(p, 128 * 4);
  float*    ve2   = (float*)carve(p, 128 * 4);
  _Float16* Wpk1  = (_Float16*)carve(p, 16384 * 2);
  _Float16* Wpk2  = (_Float16*)carve(p, 16384 * 2);

  // ---- shared preprocessing (both layers) ----
  k_vec<<<1, 128, 0, stream>>>(We1, ae1, We2, ae2, ve1, ve2);
  k_pack<<<64, 256, 0, stream>>>(W1, Wpk1);
  k_pack<<<64, 256, 0, stream>>>(W2, Wpk2);
  k_zero_pre<<<(NN + 255) / 256, 256, 0, stream>>>(deg, es1, es2);
  k_edge_dot<<<EE / 8, 256, 0, stream>>>(ea, ei, ve1, ve2, ev1, ev2, deg, es1, es2);

  dim3 gblk(32, 4);
  int ggrid = (NTILES + 3) / 4;
  int tgrid = (TT + 255) / 256;
  int sgrid = (TT + 7) / 8;

  // ---- layer 1 ----
  k_gemm<<<ggrid, gblk, 0, stream>>>(x, Wpk1, h);
  k_asad<<<(NN + 7) / 8, 256, 0, stream>>>(h, a1s, a1d, as_, ad_);
  k_init_layer<<<(NN * CC + 255) / 256, 256, 0, stream>>>(o1, b1, menc, ssum);
  k_alpha<<<tgrid, 256, 0, stream>>>(ei, ev1, es1, deg, as_, ad_, alph, menc);
  k_expsum<<<tgrid, 256, 0, stream>>>(ei, menc, alph, ssum);
  k_scatter<<<sgrid, 256, 0, stream>>>(ei, alph, ssum, h, o1);
  k_relu<<<(NN * CC + 255) / 256, 256, 0, stream>>>(o1);

  // ---- layer 2 ----
  k_gemm<<<ggrid, gblk, 0, stream>>>(o1, Wpk2, h);
  k_asad<<<(NN + 7) / 8, 256, 0, stream>>>(h, a2s, a2d, as_, ad_);
  k_init_layer<<<(NN * CC + 255) / 256, 256, 0, stream>>>(out, b2, menc, ssum);
  k_alpha<<<tgrid, 256, 0, stream>>>(ei, ev2, es2, deg, as_, ad_, alph, menc);
  k_expsum<<<tgrid, 256, 0, stream>>>(ei, menc, alph, ssum);
  k_scatter<<<sgrid, 256, 0, stream>>>(ei, alph, ssum, h, out);
}
